// DbrxFFN_40492951667586
// MI455X (gfx1250) — matrix operand order repaired and checked by reading, not compiled, not run on hardware
//
#include <hip/hip_runtime.h>

typedef __bf16 bf16_t;
typedef __attribute__((ext_vector_type(16))) __bf16 bf16x16;
typedef __attribute__((ext_vector_type(8)))  __bf16 bf16x8;
typedef __attribute__((ext_vector_type(4)))  __bf16 bf16x4;
typedef __attribute__((ext_vector_type(8)))  float  f32x8;
typedef __attribute__((ext_vector_type(4)))  int    v4i;

#define AS1 __attribute__((address_space(1)))
#define AS3 __attribute__((address_space(3)))

#define D_MODEL 2048
#define FFN     2048
#define NEXP    16
#define TOPK    4
#define NTOK    8192
#define NPAIR   (NTOK * TOPK)

#define BM 128
#define BN 64
#define BK 32
#define LDA 40   /* BK + 8 pad (row stride 80 B -> 16B-aligned fragment bases) */
#define LDB 40

// ---- optional gfx1250 async global->LDS path (probe via __has_builtin) ----
#if defined(__has_builtin)
#  if __has_builtin(__builtin_amdgcn_global_load_async_to_lds_b128)
#    define HAVE_ASYNC_LDS 1
#  endif
#endif
#ifndef HAVE_ASYNC_LDS
#  define HAVE_ASYNC_LDS 0
#endif

#if HAVE_ASYNC_LDS
#  if __has_builtin(__builtin_amdgcn_s_wait_asynccnt)
#    define ASYNC_WAIT() __builtin_amdgcn_s_wait_asynccnt(0)
#  else
#    define ASYNC_WAIT() asm volatile("s_wait_asynccnt 0x0" ::: "memory")
#  endif
#else
#  define ASYNC_WAIT() ((void)0)
#endif

// ---------- helpers ----------

__device__ __forceinline__ float f4c(const float4& v, int c) {
  const float a[4] = {v.x, v.y, v.z, v.w};
  return a[c];                      // c is always an unrolled constant
}

// build a 16-element bf16 fragment from two 16-byte LDS reads
__device__ __forceinline__ bf16x16 ld_frag16(const bf16_t* p0, const bf16_t* p1) {
  bf16x8 lo = *(const bf16x8*)p0;
  bf16x8 hi = *(const bf16x8*)p1;
  bf16x16 r;
#pragma unroll
  for (int i = 0; i < 8; ++i) { r[i] = lo[i]; r[i + 8] = hi[i]; }
  return r;
}

// ---------- kernel 0: zero output + expert counters ----------

__global__ void k_zero(float* __restrict__ out, int n, unsigned* __restrict__ counts) {
  int i = blockIdx.x * blockDim.x + threadIdx.x;
  if (i < n) out[i] = 0.0f;
  if (blockIdx.x == 0 && threadIdx.x < NEXP) counts[threadIdx.x] = 0u;
}

// ---------- kernel 1: router (one wave32 per token) ----------

__global__ __launch_bounds__(256) void k_router(const float* __restrict__ x,
                                                const float* __restrict__ wr,
                                                int* __restrict__ topk_e,
                                                float* __restrict__ topk_w) {
  const int lane = threadIdx.x & 31;
  const int wave = threadIdx.x >> 5;
  const int t = blockIdx.x * (blockDim.x >> 5) + wave;
  if (t >= NTOK) return;
  const float* xr = x + (size_t)t * D_MODEL;

  float acc[NEXP];
#pragma unroll
  for (int e = 0; e < NEXP; ++e) acc[e] = 0.0f;

  for (int d = lane; d < D_MODEL; d += 32) {
    float xv = xr[d];
#pragma unroll
    for (int e = 0; e < NEXP; ++e) acc[e] += xv * wr[e * D_MODEL + d];
  }
#pragma unroll
  for (int e = 0; e < NEXP; ++e) {
#pragma unroll
    for (int off = 16; off >= 1; off >>= 1) acc[e] += __shfl_xor(acc[e], off, 32);
  }
  float mx = acc[0];
#pragma unroll
  for (int e = 1; e < NEXP; ++e) mx = fmaxf(mx, acc[e]);
  float p[NEXP];
  float sum = 0.0f;
#pragma unroll
  for (int e = 0; e < NEXP; ++e) { p[e] = __expf(acc[e] - mx); sum += p[e]; }
  float inv = 1.0f / sum;
#pragma unroll
  for (int e = 0; e < NEXP; ++e) p[e] *= inv;

  int ids[TOPK]; float ws[TOPK];
  unsigned selmask = 0u;
  float wsum = 0.0f;
#pragma unroll
  for (int k = 0; k < TOPK; ++k) {
    float best = -1.0f; int bi = 0;
#pragma unroll
    for (int e = 0; e < NEXP; ++e) {
      bool free_e = ((selmask >> e) & 1u) == 0u;
      if (free_e && p[e] > best) { best = p[e]; bi = e; }
    }
    selmask |= (1u << bi);
    ids[k] = bi; ws[k] = best; wsum += best;
  }
  float wn = 1.0f / wsum;
  if (lane == 0) {
#pragma unroll
    for (int k = 0; k < TOPK; ++k) {
      topk_e[t * TOPK + k] = ids[k];
      topk_w[t * TOPK + k] = ws[k] * wn;
    }
  }
}

// ---------- kernel 2: bucket (token,slot) pairs per expert ----------

__global__ void k_bucket(const int* __restrict__ topk_e,
                         unsigned* __restrict__ counts,
                         int* __restrict__ bucket) {
  int i = blockIdx.x * blockDim.x + threadIdx.x;   // i = t*4 + slot
  if (i >= NPAIR) return;
  int e = topk_e[i];
  unsigned pos = atomicAdd(&counts[e], 1u);
  bucket[e * NTOK + (int)pos] = i;
}

// ---------- kernel 3: grouped gate/up GEMM + SiLU fuse -> H (bf16) ----------

__global__ __launch_bounds__(256) void k_gateup(const float* __restrict__ x,
                                                const float* __restrict__ w_gate,
                                                const float* __restrict__ w_up,
                                                const int* __restrict__ bucket,
                                                const unsigned* __restrict__ counts,
                                                bf16_t* __restrict__ H) {
  const int e = blockIdx.z;
  const int cnt = (int)counts[e];
  const int m0 = blockIdx.y * BM;
  if (m0 >= cnt) return;                    // sparse dispatch: skip empty tiles
  const int n0 = blockIdx.x * BN;

  __shared__ bf16_t sA[2][BM][LDA];
  __shared__ bf16_t sBg[2][BN][LDB];
  __shared__ bf16_t sBu[2][BN][LDB];
  __shared__ int rowsrc[BM];

  const int tid = threadIdx.x;
  const int lane = tid & 31;
  const int wave = tid >> 5;
  const int wm = (wave >> 1) * 32;          // 4 waves over M
  const int wn = (wave & 1) * 32;           // 2 waves over N
  const int khalf = lane >> 4;
  const int l15 = lane & 15;

  if (tid < BM) {
    int gr = m0 + tid;
    rowsrc[tid] = (gr < cnt) ? bucket[e * NTOK + gr] : -1;
  }
  __syncthreads();

  // B staging: thread owns 4k x 4n block of gate (tid<128) or up (tid>=128)
  const int bmat = tid >> 7;
  const int bb = tid & 127;
  const int bn = (bb & 15) << 2;            // 0..60
  const int bk = (bb >> 4) << 2;            // 0..28
  const float* __restrict__ wsrc = bmat ? w_up : w_gate;
  const size_t wbase = (size_t)e * D_MODEL * FFN;

  f32x8 accg[2][2] = {};
  f32x8 accu[2][2] = {};

  float4 rA[4];
  float4 rB[4];

  auto load_regs = [&](int kt) {
#pragma unroll
    for (int j = 0; j < 4; ++j) {
      int id = tid + j * 256;
      int r = id >> 3, kc = (id & 7) << 2;
      int src = rowsrc[r];
      if (src >= 0)
        rA[j] = *(const float4*)(x + (size_t)(src >> 2) * D_MODEL + kt + kc);
      else
        rA[j] = make_float4(0.f, 0.f, 0.f, 0.f);
    }
#pragma unroll
    for (int r = 0; r < 4; ++r)
      rB[r] = *(const float4*)(wsrc + wbase + (size_t)(kt + bk + r) * FFN + n0 + bn);
  };
  auto store_lds = [&](int buf) {
#pragma unroll
    for (int j = 0; j < 4; ++j) {
      int id = tid + j * 256;
      int r = id >> 3, kc = (id & 7) << 2;
      bf16x4 v;
      v[0] = (bf16_t)rA[j].x; v[1] = (bf16_t)rA[j].y;
      v[2] = (bf16_t)rA[j].z; v[3] = (bf16_t)rA[j].w;
      *(bf16x4*)&sA[buf][r][kc] = v;        // packed 8B ds_store
    }
    bf16_t (*sB)[LDB] = bmat ? sBu[buf] : sBg[buf];
#pragma unroll
    for (int c = 0; c < 4; ++c) {           // 4x4 register transpose -> 8B stores
      bf16x4 v;
      v[0] = (bf16_t)f4c(rB[0], c); v[1] = (bf16_t)f4c(rB[1], c);
      v[2] = (bf16_t)f4c(rB[2], c); v[3] = (bf16_t)f4c(rB[3], c);
      *(bf16x4*)&sB[bn + c][bk] = v;
    }
  };

  load_regs(0);
  store_lds(0);
  __syncthreads();

  const int NIT = D_MODEL / BK;
  for (int it = 0; it < NIT; ++it) {
    const int cur = it & 1;
    const bool more = (it + 1) < NIT;
    if (more) {
      load_regs((it + 1) * BK);             // global loads in flight over WMMAs
      if (it + 2 < NIT)
        __builtin_prefetch(wsrc + wbase + (size_t)((it + 2) * BK + bk) * FFN + n0 + bn, 0, 1);
    }

    bf16x16 a[2], bg[2], bu[2];
#pragma unroll
    for (int i = 0; i < 2; ++i) {
      const bf16_t* row = &sA[cur][wm + i * 16 + l15][0];
      a[i] = ld_frag16(row + khalf * 8, row + 16 + khalf * 8);
    }
#pragma unroll
    for (int jj = 0; jj < 2; ++jj) {
      const bf16_t* cg = &sBg[cur][wn + jj * 16 + l15][0];
      const bf16_t* cu = &sBu[cur][wn + jj * 16 + l15][0];
      bg[jj] = ld_frag16(cg + khalf * 8, cg + 16 + khalf * 8);
      bu[jj] = ld_frag16(cu + khalf * 8, cu + 16 + khalf * 8);
    }
#pragma unroll
    for (int i = 0; i < 2; ++i)
#pragma unroll
      for (int jj = 0; jj < 2; ++jj) {
        accg[i][jj] = __builtin_amdgcn_wmma_f32_16x16x32_bf16(
            false, a[i], false, bg[jj], (short)0, accg[i][jj], false, false);
        accu[i][jj] = __builtin_amdgcn_wmma_f32_16x16x32_bf16(
            false, a[i], false, bu[jj], (short)0, accu[i][jj], false, false);
      }

    if (more) store_lds(1 - cur);
    __syncthreads();
  }

  // epilogue: h = silu(g) * u, scatter bf16 rows to H[pair][F]
#pragma unroll
  for (int i = 0; i < 2; ++i)
#pragma unroll
    for (int jj = 0; jj < 2; ++jj)
#pragma unroll
      for (int v = 0; v < 8; ++v) {
        int r = wm + i * 16 + v + 8 * khalf;    // C/D: row = vgpr + 8*lane[4]
        int c = wn + jj * 16 + l15;             // col = lane&15
        int src = rowsrc[r];
        if (src >= 0) {
          float g = accg[i][jj][v];
          float u = accu[i][jj][v];
          float h = (g / (1.0f + __expf(-g))) * u;
          H[(size_t)src * FFN + n0 + c] = (bf16_t)h;
        }
      }
}

// ---------- kernel 4: grouped down GEMM + weighted scatter-add ----------

__global__ __launch_bounds__(256) void k_down(const bf16_t* __restrict__ H,
                                              const float* __restrict__ w_down,
                                              const int* __restrict__ bucket,
                                              const unsigned* __restrict__ counts,
                                              const float* __restrict__ topk_w,
                                              float* __restrict__ out) {
  const int e = blockIdx.z;
  const int cnt = (int)counts[e];
  const int m0 = blockIdx.y * BM;
  if (m0 >= cnt) return;
  const int n0 = blockIdx.x * BN;

  __shared__ bf16_t sA[2][BM][LDA];
  __shared__ bf16_t sB[2][BN][LDB];
  __shared__ int rowsrc[BM];

  const int tid = threadIdx.x;
  const int lane = tid & 31;
  const int wave = tid >> 5;
  const int wm = (wave >> 1) * 32;
  const int wn = (wave & 1) * 32;
  const int khalf = lane >> 4;
  const int l15 = lane & 15;

  if (tid < BM) {
    int gr = m0 + tid;
    rowsrc[tid] = (gr < cnt) ? bucket[e * NTOK + gr] : -1;
  }
  __syncthreads();

  // B staging: thread owns 4k x 2n block of w_down
  const int bn = (tid & 31) << 1;           // 0..62
  const int bk = (tid >> 5) << 2;           // 0..28
  const size_t wbase = (size_t)e * FFN * D_MODEL;

  f32x8 acc[2][2] = {};

  bf16x8 rA2[2];
  float2 rB2[4];

#if HAVE_ASYNC_LDS
  auto stage_A_async = [&](int buf, int kt) {
#pragma unroll
    for (int j = 0; j < 2; ++j) {
      int id = tid + j * 256;
      int r = id >> 2, kc = (id & 3) << 3;
      int src = rowsrc[r];
      // invalid rows copy row 0 (harmless: their outputs are masked)
      const bf16_t* gp = H + (size_t)(src >= 0 ? src : 0) * FFN + kt + kc;
      __builtin_amdgcn_global_load_async_to_lds_b128(
          (AS1 v4i*)gp,
          (AS3 v4i*)&sA[buf][r][kc], 0, 0);
    }
  };
#endif

  auto load_regs = [&](int kt) {
#if !HAVE_ASYNC_LDS
#pragma unroll
    for (int j = 0; j < 2; ++j) {
      int id = tid + j * 256;
      int r = id >> 2, kc = (id & 3) << 3;
      int src = rowsrc[r];
      if (src >= 0) {
        rA2[j] = *(const bf16x8*)(H + (size_t)src * FFN + kt + kc);
      } else {
#pragma unroll
        for (int q = 0; q < 8; ++q) rA2[j][q] = (bf16_t)0.0f;
      }
    }
#endif
#pragma unroll
    for (int r = 0; r < 4; ++r)
      rB2[r] = *(const float2*)(w_down + wbase + (size_t)(kt + bk + r) * D_MODEL + n0 + bn);
  };
  auto store_lds = [&](int buf) {
#if !HAVE_ASYNC_LDS
#pragma unroll
    for (int j = 0; j < 2; ++j) {
      int id = tid + j * 256;
      int r = id >> 2, kc = (id & 3) << 3;
      *(bf16x8*)&sA[buf][r][kc] = rA2[j];
    }
#endif
#pragma unroll
    for (int c = 0; c < 2; ++c) {
      bf16x4 v;
      v[0] = (bf16_t)(c ? rB2[0].y : rB2[0].x);
      v[1] = (bf16_t)(c ? rB2[1].y : rB2[1].x);
      v[2] = (bf16_t)(c ? rB2[2].y : rB2[2].x);
      v[3] = (bf16_t)(c ? rB2[3].y : rB2[3].x);
      *(bf16x4*)&sB[buf][bn + c][bk] = v;
    }
  };

#if HAVE_ASYNC_LDS
  stage_A_async(0, 0);
#endif
  load_regs(0);
  store_lds(0);
  ASYNC_WAIT();
  __syncthreads();

  const int NIT = FFN / BK;
  for (int it = 0; it < NIT; ++it) {
    const int cur = it & 1;
    const bool more = (it + 1) < NIT;
    if (more) {
#if HAVE_ASYNC_LDS
      stage_A_async(1 - cur, (it + 1) * BK);
#endif
      load_regs((it + 1) * BK);
      if (it + 2 < NIT)
        __builtin_prefetch(w_down + wbase + (size_t)((it + 2) * BK + bk) * D_MODEL + n0 + bn, 0, 1);
    }

    bf16x16 a[2], b[2];
#pragma unroll
    for (int i = 0; i < 2; ++i) {
      const bf16_t* row = &sA[cur][wm + i * 16 + l15][0];
      a[i] = ld_frag16(row + khalf * 8, row + 16 + khalf * 8);
    }
#pragma unroll
    for (int jj = 0; jj < 2; ++jj) {
      const bf16_t* cb = &sB[cur][wn + jj * 16 + l15][0];
      b[jj] = ld_frag16(cb + khalf * 8, cb + 16 + khalf * 8);
    }
#pragma unroll
    for (int i = 0; i < 2; ++i)
#pragma unroll
      for (int jj = 0; jj < 2; ++jj)
        acc[i][jj] = __builtin_amdgcn_wmma_f32_16x16x32_bf16(
            false, a[i], false, b[jj], (short)0, acc[i][jj], false, false);

    if (more) store_lds(1 - cur);
    ASYNC_WAIT();
    __syncthreads();
  }

  // epilogue: out[token] += router_weight * acc
#pragma unroll
  for (int i = 0; i < 2; ++i)
#pragma unroll
    for (int jj = 0; jj < 2; ++jj)
#pragma unroll
      for (int v = 0; v < 8; ++v) {
        int r = wm + i * 16 + v + 8 * khalf;
        int c = wn + jj * 16 + l15;
        int src = rowsrc[r];
        if (src >= 0) {
          float coeff = topk_w[src];
          int tok = src >> 2;
          atomicAdd(out + (size_t)tok * D_MODEL + n0 + c, coeff * acc[i][jj][v]);
        }
      }
}

// ---------- host launcher ----------

extern "C" void kernel_launch(void* const* d_in, const int* in_sizes, int n_in,
                              void* d_out, int out_size, void* d_ws, size_t ws_size,
                              hipStream_t stream) {
  const float* x        = (const float*)d_in[0];
  const float* w_router = (const float*)d_in[1];
  const float* w_gate   = (const float*)d_in[2];
  const float* w_up     = (const float*)d_in[3];
  const float* w_down   = (const float*)d_in[4];
  float* out = (float*)d_out;

  char* ws = (char*)d_ws;
  unsigned* counts = (unsigned*)ws;                        //       64 B
  int*      topk_e = (int*)(ws + 256);                     //  131072 B
  float*    topk_w = (float*)(ws + 256 + 131072);          //  131072 B
  int*      bucket = (int*)(ws + 256 + 262144);            //  524288 B
  bf16_t*   H      = (bf16_t*)(ws + 1048576);              //  134 MB (NPAIR * FFN * 2)

  int nOut = NTOK * D_MODEL;
  k_zero<<<(nOut + 255) / 256, 256, 0, stream>>>(out, nOut, counts);
  k_router<<<NTOK / 8, 256, 0, stream>>>(x, w_router, topk_e, topk_w);
  k_bucket<<<NPAIR / 256, 256, 0, stream>>>(topk_e, counts, bucket);

  dim3 g1(FFN / BN, NTOK / BM, NEXP);
  k_gateup<<<g1, 256, 0, stream>>>(x, w_gate, w_up, bucket, counts, H);

  dim3 g2(D_MODEL / BN, NTOK / BM, NEXP);
  k_down<<<g2, 256, 0, stream>>>(H, w_down, bucket, counts, topk_w, out);
}
